// SelfAttentionBlock_4904852652309
// MI455X (gfx1250) — matrix operand order repaired and checked by reading, not compiled, not run on hardware
//
#include <hip/hip_runtime.h>

typedef __attribute__((ext_vector_type(16))) __bf16 v16bf;
typedef __attribute__((ext_vector_type(8)))  float  v8f;

constexpr int Bc = 2, Tc = 2048, Dc = 2048, Hc = 16, DHc = 128, FFNc = 8192;
constexpr float EPSc = 1e-5f;

// -------- s_wait_asynccnt helper (ASYNCcnt <= N) --------
template <int N> __device__ __forceinline__ void wait_asynccnt() {
  if constexpr (N == 0)      asm volatile("s_wait_asynccnt 0x0" ::: "memory");
  else if constexpr (N == 2) asm volatile("s_wait_asynccnt 0x2" ::: "memory");
  else if constexpr (N == 4) asm volatile("s_wait_asynccnt 0x4" ::: "memory");
}

// -------- async global -> LDS copy of 32 bytes (2x b128), ASYNCcnt-tracked --------
__device__ __forceinline__ void async_copy32(unsigned lds_addr, const void* gptr) {
  asm volatile("global_load_async_to_lds_b128 %0, %1, off"
               :: "v"(lds_addr), "v"(gptr) : "memory");
  asm volatile("global_load_async_to_lds_b128 %0, %1, off offset:16"
               :: "v"(lds_addr), "v"(gptr) : "memory");
}

// ---------------- elementwise f32 -> bf16 convert ----------------
__global__ void cvt_bf16_kernel(const float* __restrict__ src,
                                __bf16* __restrict__ dst, size_t n) {
  size_t i = (size_t)blockIdx.x * blockDim.x + threadIdx.x;
  size_t stride = (size_t)gridDim.x * blockDim.x;
  for (; i < n; i += stride) dst[i] = (__bf16)src[i];
}

// ---------------- LayerNorm: f32 in -> bf16 out (D == 2048) ----------------
__global__ void ln_bf16_kernel(const float* __restrict__ x,
                               const float* __restrict__ scale,
                               const float* __restrict__ offset,
                               __bf16* __restrict__ out) {
  int row = blockIdx.x;
  int tid = threadIdx.x;
  const float* xr = x + (size_t)row * Dc;
  __bf16* orow = out + (size_t)row * Dc;
  float vals[8];
  float lsum = 0.f, lsq = 0.f;
#pragma unroll
  for (int i = 0; i < 8; ++i) {
    int c = tid + i * 256;
    float v = xr[c];
    vals[i] = v;
    lsum += v; lsq += v * v;
  }
  __shared__ float s0[256], s1[256];
  s0[tid] = lsum; s1[tid] = lsq;
  __syncthreads();
  for (int off = 128; off > 0; off >>= 1) {
    if (tid < off) { s0[tid] += s0[tid + off]; s1[tid] += s1[tid + off]; }
    __syncthreads();
  }
  float mu = s0[0] * (1.f / Dc);
  float var = s1[0] * (1.f / Dc) - mu * mu;
  float rstd = rsqrtf(var + EPSc);
#pragma unroll
  for (int i = 0; i < 8; ++i) {
    int c = tid + i * 256;
    float v = (vals[i] - mu) * rstd * scale[c] + offset[c];
    orow[c] = (__bf16)v;
  }
}

// ---------------- masked softmax over last dim (T == 2048), in place ----------------
__global__ void softmax_mask_kernel(float* __restrict__ attn,
                                    const unsigned char* __restrict__ mask) {
  int row = blockIdx.x;                  // B*H*T rows
  int tid = threadIdx.x;
  int b = row / (Hc * Tc);
  float* pr = attn + (size_t)row * Tc;
  const unsigned char* mrow = mask + (size_t)b * Tc;
  float vals[8];
  float lmax = -3.0e38f;
#pragma unroll
  for (int i = 0; i < 8; ++i) {
    int c = tid + i * 256;
    float v = pr[c];
    v = mrow[c] ? v : -1.0e30f;
    vals[i] = v;
    lmax = fmaxf(lmax, v);
  }
  __shared__ float sm[256];
  sm[tid] = lmax; __syncthreads();
  for (int off = 128; off > 0; off >>= 1) {
    if (tid < off) sm[tid] = fmaxf(sm[tid], sm[tid + off]);
    __syncthreads();
  }
  float rmax = sm[0];
  __syncthreads();
  float lsum = 0.f;
#pragma unroll
  for (int i = 0; i < 8; ++i) { vals[i] = expf(vals[i] - rmax); lsum += vals[i]; }
  sm[tid] = lsum; __syncthreads();
  for (int off = 128; off > 0; off >>= 1) {
    if (tid < off) sm[tid] += sm[tid + off];
    __syncthreads();
  }
  float inv = 1.f / sm[0];
#pragma unroll
  for (int i = 0; i < 8; ++i) { int c = tid + i * 256; pr[c] = vals[i] * inv; }
}

// ---------------- WMMA bf16 GEMM, 128x128 block tile, BK=32, double-buffered ----------------
// C[M,N] = act(scale * A[M,K] @ B + bias) + res ; per-z (batch,head) offsets.
// A_F32   : A global buffer is f32, converted to bf16 while staging (VGPR path).
// TRANS_B : B global buffer is [N,K] row-major (computes A @ B^T, async path).
// Contiguous bf16 tiles (A when !A_F32, B when TRANS_B) are staged with
// global_load_async_to_lds_b128 (ASYNCcnt) and double-buffered against WMMA.
template <bool A_F32, bool TRANS_B, bool OUT_BF16, bool HAS_BIAS, bool DO_GELU, bool HAS_RES>
__global__ __launch_bounds__(256) void gemm_wmma_kernel(
    const void* __restrict__ Ag, const __bf16* __restrict__ Bg,
    float* __restrict__ Cf, __bf16* __restrict__ Cb,
    const float* __restrict__ bias, const float* __restrict__ res,
    int K, int lda, int ldb, int ldc,
    long long aB, long long aH, long long bB, long long bH,
    long long cB, long long cH, int Hh, float scl) {
  __shared__ alignas(16) __bf16 As[2][128][40];   // [buf][m][k], +8 bf16 pad (80B row = 5x16B)
  __shared__ alignas(16) __bf16 Bs[2][128][40];   // [buf][n][k]

  constexpr int PEND = (A_F32 ? 0 : 2) + (TRANS_B ? 2 : 0);  // async ops/thread/tile

  int z = blockIdx.z;
  int bi = z / Hh, hi = z % Hh;
  const __bf16* Abf = (const __bf16*)Ag + (size_t)bi * aB + (size_t)hi * aH;
  const float*  Af  = (const float*)Ag + (size_t)bi * aB + (size_t)hi * aH;
  const __bf16* Bp  = Bg + (size_t)bi * bB + (size_t)hi * bH;
  size_t cOff = (size_t)bi * cB + (size_t)hi * cH;

  int m0 = blockIdx.y * 128;
  int n0 = blockIdx.x * 128;
  int tid = threadIdx.x;
  int lane = tid & 31;
  int wave = tid >> 5;
  int wm = (wave >> 1) * 32;     // wave tile: 32 rows x 64 cols
  int wn = (wave & 1) * 64;
  int half = lane >> 4;
  int l16 = lane & 15;

  int r   = tid >> 1;            // contiguous staging: row index (0..127)
  int c16 = (tid & 1) * 16;      // contiguous staging: 16-element K half
  int kk   = tid & 31;           // NN-B staging: K row (0..31)
  int nseg = tid >> 5;           // NN-B staging: 16-wide N segment (0..7)

  float aregs[16];               // A_F32 prefetch registers
  uint4 breg0, breg1;            // NN-B prefetch registers (16 bf16)

  // issue tile loads: async ops straight to LDS, or global->VGPR prefetch
  auto issue_tile = [&](int k0, int buf) {
    if constexpr (A_F32) {
      const float* src = Af + (size_t)(m0 + r) * lda + k0 + c16;
#pragma unroll
      for (int i = 0; i < 16; ++i) aregs[i] = src[i];
    } else {
      unsigned lds = (unsigned)(size_t)&As[buf][r][c16];
      async_copy32(lds, Abf + (size_t)(m0 + r) * lda + k0 + c16);
    }
    if constexpr (TRANS_B) {
      unsigned lds = (unsigned)(size_t)&Bs[buf][r][c16];
      async_copy32(lds, Bp + (size_t)(n0 + r) * ldb + k0 + c16);
    } else {
      const uint4* src =
          (const uint4*)(Bp + (size_t)(k0 + kk) * ldb + n0 + nseg * 16);
      breg0 = src[0];
      breg1 = src[1];
    }
  };

  v8f acc[2][4];
#pragma unroll
  for (int i = 0; i < 2; ++i)
#pragma unroll
    for (int j = 0; j < 4; ++j) acc[i][j] = (v8f)(0.0f);

  issue_tile(0, 0);
  int ntiles = K >> 5;
  for (int t = 0; t < ntiles; ++t) {
    int buf = t & 1;
    // ---- write VGPR-prefetched pieces of tile t into LDS ----
    if constexpr (A_F32) {
      unsigned* dst = (unsigned*)&As[buf][r][c16];
#pragma unroll
      for (int i = 0; i < 8; ++i) {
        union { __bf16 b2[2]; unsigned u; } cv;
        cv.b2[0] = (__bf16)aregs[2 * i];
        cv.b2[1] = (__bf16)aregs[2 * i + 1];
        dst[i] = cv.u;
      }
    }
    if constexpr (!TRANS_B) {
      union { uint4 u4[2]; __bf16 h[16]; } cv;
      cv.u4[0] = breg0; cv.u4[1] = breg1;
#pragma unroll
      for (int i = 0; i < 16; ++i) Bs[buf][nseg * 16 + i][kk] = cv.h[i];
    }
    // ---- prefetch tile t+1 (async into other buffer / VGPRs) ----
    bool has_next = (t + 1 < ntiles);
    if (has_next) issue_tile((t + 1) << 5, buf ^ 1);
    if constexpr (PEND > 0) {
      if (has_next) wait_asynccnt<PEND>();  // tile t's async ops have landed
      else          wait_asynccnt<0>();
    }
    __syncthreads();

    // ---- fragments per ISA layouts ----
    v16bf afr[2], bfr[4];
#pragma unroll
    for (int i = 0; i < 2; ++i) {
      const unsigned* arow = (const unsigned*)&As[buf][wm + i * 16 + l16][0];
      union { v16bf v; unsigned u[8]; } tt;
#pragma unroll
      for (int q = 0; q < 8; ++q)             // K = (q/4)*16 + half*8 + (q%2)*2... = (q>>2)*16+half*8+(q&3)*2
        tt.u[q] = arow[(q >> 2) * 8 + half * 4 + (q & 3)];
      afr[i] = tt.v;
    }
#pragma unroll
    for (int j = 0; j < 4; ++j) {
      const unsigned* brow = (const unsigned*)&Bs[buf][wn + j * 16 + l16][0];
      union { v16bf v; unsigned u[8]; } tt;
#pragma unroll
      for (int q = 0; q < 8; ++q)             // K = (q>>2)*16+half*8+(q&3)*2
        tt.u[q] = brow[(q >> 2) * 8 + half * 4 + (q & 3)];
      bfr[j] = tt.v;
    }
#pragma unroll
    for (int i = 0; i < 2; ++i)
#pragma unroll
      for (int j = 0; j < 4; ++j)
        acc[i][j] = __builtin_amdgcn_wmma_f32_16x16x32_bf16(
            false, afr[i], false, bfr[j], (short)0, acc[i][j], false, false);
    __syncthreads();
  }

  // ---- epilogue: C row = r + 8*half, col = lane&15 ----
#pragma unroll
  for (int i = 0; i < 2; ++i) {
#pragma unroll
    for (int j = 0; j < 4; ++j) {
      int col = n0 + wn + j * 16 + l16;
      float bval = HAS_BIAS ? bias[col] : 0.f;
#pragma unroll
      for (int rr = 0; rr < 8; ++rr) {
        int row = m0 + wm + i * 16 + rr + half * 8;
        float x = acc[i][j][rr] * scl + bval;
        if (DO_GELU) x = 0.5f * x * (1.f + erff(x * 0.70710678118654752f));
        size_t idx = cOff + (size_t)row * ldc + col;
        if (HAS_RES) x += res[idx];
        if (OUT_BF16) Cb[idx] = (__bf16)x;
        else          Cf[idx] = x;
      }
    }
  }
}

extern "C" void kernel_launch(void* const* d_in, const int* in_sizes, int n_in,
                              void* d_out, int out_size, void* d_ws, size_t ws_size,
                              hipStream_t stream) {
  (void)in_sizes; (void)n_in; (void)out_size; (void)ws_size;
  const float* x    = (const float*)d_in[0];
  const unsigned char* amask = (const unsigned char*)d_in[1];
  const float* ln1s = (const float*)d_in[2];
  const float* ln1o = (const float*)d_in[3];
  const float* ln2s = (const float*)d_in[4];
  const float* ln2o = (const float*)d_in[5];
  const float* wq = (const float*)d_in[6];  const float* bq = (const float*)d_in[7];
  const float* wk = (const float*)d_in[8];  const float* bk = (const float*)d_in[9];
  const float* wv = (const float*)d_in[10]; const float* bv = (const float*)d_in[11];
  const float* wo = (const float*)d_in[12]; const float* bo = (const float*)d_in[13];
  const float* w1 = (const float*)d_in[14]; const float* b1 = (const float*)d_in[15];
  const float* w2 = (const float*)d_in[16]; const float* b2 = (const float*)d_in[17];

  float* out  = (float*)d_out;
  float* attn = out + (size_t)Bc * Tc * Dc;   // [B,H,T,T] f32 (second output)

  const size_t MD = (size_t)Bc * Tc * Dc;     // 4096*2048
  const size_t DD = (size_t)Dc * Dc;
  const size_t DF = (size_t)Dc * FFNc;

  char* ws = (char*)d_ws;
  auto carve = [&](size_t bytes) {
    char* p = ws; ws += (bytes + 255) & ~(size_t)255; return p;
  };
  __bf16* h1  = (__bf16*)carve(MD * 2);
  __bf16* wqb = (__bf16*)carve(DD * 2);
  __bf16* wkb = (__bf16*)carve(DD * 2);
  __bf16* wvb = (__bf16*)carve(DD * 2);
  __bf16* wob = (__bf16*)carve(DD * 2);
  __bf16* w1b = (__bf16*)carve(DF * 2);
  __bf16* w2b = (__bf16*)carve(DF * 2);
  __bf16* qb  = (__bf16*)carve(MD * 2);
  __bf16* kb  = (__bf16*)carve(MD * 2);
  __bf16* vb  = (__bf16*)carve(MD * 2);
  __bf16* ctx = (__bf16*)carve(MD * 2);
  float*  x2  = (float*) carve(MD * 4);
  __bf16* h2  = (__bf16*)carve(MD * 2);
  __bf16* f1  = (__bf16*)carve((size_t)Bc * Tc * FFNc * 2);

  dim3 blk(256);
  const float inv_sqrt_dh = 0.08838834764831845f;   // 1/sqrt(128)

  // weights -> bf16
  cvt_bf16_kernel<<<4096, blk, 0, stream>>>(wq, wqb, DD);
  cvt_bf16_kernel<<<4096, blk, 0, stream>>>(wk, wkb, DD);
  cvt_bf16_kernel<<<4096, blk, 0, stream>>>(wv, wvb, DD);
  cvt_bf16_kernel<<<4096, blk, 0, stream>>>(wo, wob, DD);
  cvt_bf16_kernel<<<8192, blk, 0, stream>>>(w1, w1b, DF);
  cvt_bf16_kernel<<<8192, blk, 0, stream>>>(w2, w2b, DF);

  // LN1
  ln_bf16_kernel<<<Bc * Tc, blk, 0, stream>>>(x, ln1s, ln1o, h1);

  // Q,K,V projections: [4096,2048] = h1 @ W + b  (bf16 out)
  gemm_wmma_kernel<false, false, true, true, false, false>
      <<<dim3(Dc / 128, (Bc * Tc) / 128, 1), blk, 0, stream>>>(
          h1, wqb, nullptr, qb, bq, nullptr, Dc, Dc, Dc, Dc,
          0, 0, 0, 0, 0, 0, 1, 1.f);
  gemm_wmma_kernel<false, false, true, true, false, false>
      <<<dim3(Dc / 128, (Bc * Tc) / 128, 1), blk, 0, stream>>>(
          h1, wkb, nullptr, kb, bk, nullptr, Dc, Dc, Dc, Dc,
          0, 0, 0, 0, 0, 0, 1, 1.f);
  gemm_wmma_kernel<false, false, true, true, false, false>
      <<<dim3(Dc / 128, (Bc * Tc) / 128, 1), blk, 0, stream>>>(
          h1, wvb, nullptr, vb, bv, nullptr, Dc, Dc, Dc, Dc,
          0, 0, 0, 0, 0, 0, 1, 1.f);

  // logits[b,h] = (Q_bh @ K_bh^T) / sqrt(DH)  -> attn (f32), per (b,h)
  gemm_wmma_kernel<false, true, false, false, false, false>
      <<<dim3(Tc / 128, Tc / 128, Bc * Hc), blk, 0, stream>>>(
          qb, kb, attn, nullptr, nullptr, nullptr, DHc, Dc, Dc, Tc,
          (long long)Tc * Dc, DHc, (long long)Tc * Dc, DHc,
          (long long)Hc * Tc * Tc, (long long)Tc * Tc, Hc, inv_sqrt_dh);

  // mask + softmax in place (this region of d_out IS the attn_weights output)
  softmax_mask_kernel<<<Bc * Hc * Tc, blk, 0, stream>>>(attn, amask);

  // ctx[b, :, h, :] = P_bh @ V_bh   (A is f32 probs, converted while staging)
  gemm_wmma_kernel<true, false, true, false, false, false>
      <<<dim3(DHc / 128, Tc / 128, Bc * Hc), blk, 0, stream>>>(
          attn, vb, nullptr, ctx, nullptr, nullptr, Tc, Tc, Dc, Dc,
          (long long)Hc * Tc * Tc, (long long)Tc * Tc,
          (long long)Tc * Dc, DHc, (long long)Tc * Dc, DHc, Hc, 1.f);

  // x2 = x + ctx @ wo + bo  (f32)
  gemm_wmma_kernel<false, false, false, true, false, true>
      <<<dim3(Dc / 128, (Bc * Tc) / 128, 1), blk, 0, stream>>>(
          ctx, wob, x2, nullptr, bo, x, Dc, Dc, Dc, Dc,
          0, 0, 0, 0, 0, 0, 1, 1.f);

  // LN2
  ln_bf16_kernel<<<Bc * Tc, blk, 0, stream>>>(x2, ln2s, ln2o, h2);

  // f1 = gelu(h2 @ w1 + b1)  (bf16)
  gemm_wmma_kernel<false, false, true, true, true, false>
      <<<dim3(FFNc / 128, (Bc * Tc) / 128, 1), blk, 0, stream>>>(
          h2, w1b, nullptr, f1, b1, nullptr, Dc, Dc, FFNc, FFNc,
          0, 0, 0, 0, 0, 0, 1, 1.f);

  // out = x2 + f1 @ w2 + b2  (f32, first output)
  gemm_wmma_kernel<false, false, false, true, false, true>
      <<<dim3(Dc / 128, (Bc * Tc) / 128, 1), blk, 0, stream>>>(
          f1, w2b, out, nullptr, b2, x2, FFNc, FFNc, Dc, Dc,
          0, 0, 0, 0, 0, 0, 1, 1.f);
}
